// MultiHeadAttention_4337916969310
// MI455X (gfx1250) — compile-verified, matrix operand order checked by reading, not run on hardware
//
#include <hip/hip_runtime.h>
#include <hip/hip_bf16.h>

typedef __bf16 bf16;
typedef __attribute__((ext_vector_type(16))) __bf16 v16bf;
typedef __attribute__((ext_vector_type(8)))  float  v8f;

#define D_MODEL 1024
#define N_HEADS 16
#define DK      64
#define SEQ     2048
#define BATCH   2

#if __has_builtin(__builtin_amdgcn_global_load_async_to_lds_b128)
#define HAS_ASYNC_LDS 1
typedef int v4i __attribute__((vector_size(16)));
typedef __attribute__((address_space(1))) v4i GV4;   // global int4*
typedef __attribute__((address_space(3))) v4i LV4;   // LDS int4*
#else
#define HAS_ASYNC_LDS 0
#endif

static __device__ __forceinline__ v8f wmma_bf16(v16bf a, v16bf b, v8f c) {
  // 8 args: (neg_a, A, neg_b, B, c_mod, C, reuse_a, reuse_b)
  return __builtin_amdgcn_wmma_f32_16x16x32_bf16(false, a, false, b, (short)0, c, false, false);
}

// ---------------------------------------------------------------------------
// Elementwise f32 -> bf16 conversion (operand staging; bandwidth trivial)
// ---------------------------------------------------------------------------
__global__ __launch_bounds__(256) void cvt_f32_bf16(const float* __restrict__ src,
                                                    bf16* __restrict__ dst, int n) {
  int i = (blockIdx.x * 256 + threadIdx.x) * 4;
  if (i + 3 < n) {
    float4 v = *(const float4*)(src + i);
    dst[i + 0] = (bf16)v.x;
    dst[i + 1] = (bf16)v.y;
    dst[i + 2] = (bf16)v.z;
    dst[i + 3] = (bf16)v.w;
  } else {
    for (; i < n; ++i) dst[i] = (bf16)src[i];
  }
}

// ---------------------------------------------------------------------------
// Tiled WMMA GEMM:  C[M,N] = A[M,K] * W[N,K]^T  (both row-major bf16)
// Block: 256 threads = 8 waves arranged 4(M) x 2(N); wave computes 16x64.
// ---------------------------------------------------------------------------
template<bool OUT_F32, bool ADD_BIAS>
__global__ __launch_bounds__(256) void gemm_bf16_nt(
    const bf16* __restrict__ A, const bf16* __restrict__ W,
    const float* __restrict__ bias,
    float* __restrict__ Cf, bf16* __restrict__ Cb,
    int M, int N, int K) {
  const int lane = threadIdx.x & 31;
  const int wave = threadIdx.x >> 5;
  const int m_w  = blockIdx.y * 64  + (wave & 3) * 16;
  const int n_w  = blockIdx.x * 128 + (wave >> 2) * 64;

  v8f acc[4] = {};
  const int arow = m_w + (lane & 15);
  const int koff = (lane >> 4) * 16;   // half-lane split of K=32 fragment

  for (int k0 = 0; k0 < K; k0 += 32) {
    v16bf a = *(const v16bf*)(A + (size_t)arow * K + k0 + koff);
    if (k0 + 96 < K)
      __builtin_prefetch(A + (size_t)arow * K + k0 + 96 + koff, 0, 1); // global_prefetch_b8
    const bf16* wrow = W + (size_t)(n_w + (lane & 15)) * K + k0 + koff;
#pragma unroll
    for (int j = 0; j < 4; ++j) {
      v16bf b = *(const v16bf*)(wrow + (size_t)j * 16 * K);
      acc[j] = wmma_bf16(a, b, acc[j]);
    }
  }

  const int rbase = (lane >> 4) << 3;   // C layout: lanes 16-31 hold M+8
  const int col0  = lane & 15;
#pragma unroll
  for (int j = 0; j < 4; ++j) {
    const int col = n_w + j * 16 + col0;
#pragma unroll
    for (int r = 0; r < 8; ++r) {
      const int row = m_w + rbase + r;
      float v = acc[j][r];
      if (OUT_F32) {
        if (ADD_BIAS) v += bias[col];
        Cf[(size_t)row * N + col] = v;
      } else {
        Cb[(size_t)row * N + col] = (bf16)v;
      }
    }
  }
}

// ---------------------------------------------------------------------------
// RoPE + reshape: (B,S,D) bf16 -> (B,H,S,dk) bf16, rotate pairs (j, j+32).
// `scale` folds 1/sqrt(dk) into Q so the scores WMMA needs no epilogue scale.
// ---------------------------------------------------------------------------
__global__ __launch_bounds__(256) void rope_reshape(const bf16* __restrict__ x,
                                                    bf16* __restrict__ out,
                                                    float scale) {
  size_t idx = (size_t)blockIdx.x * 256 + threadIdx.x;   // B*S*H*32 threads
  if (idx >= (size_t)BATCH * SEQ * N_HEADS * 32) return;
  const int j = idx & 31;
  const int h = (idx >> 5) & (N_HEADS - 1);
  const size_t sb = idx >> 9;           // b*S + s
  const int s = (int)(sb & (SEQ - 1));
  const int b = (int)(sb >> 11);

  const bf16* px = x + sb * D_MODEL + h * DK;
  float x0 = (float)px[j];
  float x1 = (float)px[j + 32];
  float invf = __powf(10000.0f, -2.0f * (float)j / (float)D_MODEL);
  float ang = (float)s * invf;
  float c = __cosf(ang), sn = __sinf(ang);

  bf16* po = out + (((size_t)(b * N_HEADS + h) * SEQ) + s) * DK;
  po[j]      = (bf16)((x0 * c - x1 * sn) * scale);
  po[j + 32] = (bf16)((x1 * c + x0 * sn) * scale);
}

// ---------------------------------------------------------------------------
// V transpose: (B,S,D) bf16 -> (B,H,dk,S) bf16 so ctx B-fragments are
// contiguous 128-bit loads along the K (sequence) axis.
// ---------------------------------------------------------------------------
__global__ __launch_bounds__(256) void v_transpose(const bf16* __restrict__ x,
                                                   bf16* __restrict__ out) {
  size_t idx = (size_t)blockIdx.x * 256 + threadIdx.x;   // B*H*DK*SEQ = 2^22
  if (idx >= (size_t)BATCH * N_HEADS * DK * SEQ) return;
  const int s = (int)(idx & (SEQ - 1));
  const int d = (int)((idx >> 11) & (DK - 1));
  const int h = (int)((idx >> 17) & (N_HEADS - 1));
  const int b = (int)(idx >> 21);
  out[idx] = x[((size_t)(b * SEQ + s)) * D_MODEL + h * DK + d];
}

// ---------------------------------------------------------------------------
// Fused attention: one block per (b, h, 16-row q block).
// LDS: 16x2048 f32 score strip (128KB) + bias row (8.3KB) + q tile (2KB).
// Q staged by async-to-LDS; bias pre-loaded into the WMMA C accumulator;
// float4 softmax in LDS; attn written to HBM exactly once; ctx = P@V with
// the K reduction split across all 8 waves + LDS combine.
// ---------------------------------------------------------------------------
__global__ __launch_bounds__(256) void attn_fused(
    const bf16* __restrict__ qh, const bf16* __restrict__ kh,
    const bf16* __restrict__ vt, const float* __restrict__ rel,
    float* __restrict__ attn_out, bf16* __restrict__ ctx) {
  extern __shared__ char smem_raw[];
  float* sP    = (float*)smem_raw;        // 16 * SEQ
  float* sBias = sP + 16 * SEQ;           // 2080 (2063 used); reused as sRed
  bf16*  sQ    = (bf16*)(sBias + 2080);   // 16 * DK (32B aligned)

  const int bid = blockIdx.x;
  const int qb  = bid & 127;
  const int h   = (bid >> 7) & (N_HEADS - 1);
  const int b   = bid >> 11;
  const int bh  = b * N_HEADS + h;

  // ---- stage q tile (async DMA to LDS when available) + bias strip ----
  const bf16* qbase = qh + ((size_t)bh * SEQ + qb * 16) * DK;
#if HAS_ASYNC_LDS
  if (threadIdx.x < 128) {   // 128 lanes x 16B = 2KB tile
    __builtin_amdgcn_global_load_async_to_lds_b128(
        (GV4*)(qbase + threadIdx.x * 8),
        (LV4*)(sQ + threadIdx.x * 8), 0, 0);
  }
#if __has_builtin(__builtin_amdgcn_s_wait_asynccnt)
  __builtin_amdgcn_s_wait_asynccnt(0);
#else
  asm volatile("s_wait_asynccnt 0x0" ::: "memory");
#endif
#else
  for (int i = threadIdx.x; i < 16 * DK; i += 256) sQ[i] = qbase[i];
#endif
  for (int t = threadIdx.x; t < 2063; t += 256)
    sBias[t] = rel[(size_t)(qb * 16 + t) * N_HEADS + h];
  __syncthreads();

  const int lane = threadIdx.x & 31;
  const int wave = threadIdx.x >> 5;
  const int frow = lane & 15;
  const int fk   = (lane >> 4) * 16;
  const int rbase = (lane >> 4) << 3;

  // q fragments (A-matrix 16x32, dk=64 -> two K-halves); q pre-scaled by 1/8
  v16bf qa0 = *(const v16bf*)(sQ + frow * DK + fk);
  v16bf qa1 = *(const v16bf*)(sQ + frow * DK + 32 + fk);

  // ---- scores: each wave owns column tiles wave, wave+8, ... (128 tiles) ----
  const bf16* kbase = kh + (size_t)bh * SEQ * DK;
  for (int t = 0; t < 16; ++t) {
    const int ct = wave + t * 8;
    const bf16* krow = kbase + (size_t)(ct * 16 + frow) * DK + fk;
    v16bf kb0 = *(const v16bf*)(krow);
    v16bf kb1 = *(const v16bf*)(krow + 32);
    if (t + 1 < 16)
      __builtin_prefetch(kbase + (size_t)((ct + 8) * 16 + frow) * DK + fk, 0, 1);
    // C initialized with rel_bias: bias idx = row + 2047 - col, in [0,2062]
    const int colj  = ct * 16 + frow;
    const int bidx0 = rbase + 2047 - colj;
    v8f acc;
#pragma unroll
    for (int r = 0; r < 8; ++r) acc[r] = sBias[bidx0 + r];
    acc = wmma_bf16(qa0, kb0, acc);
    acc = wmma_bf16(qa1, kb1, acc);
#pragma unroll
    for (int r = 0; r < 8; ++r) sP[(rbase + r) * SEQ + colj] = acc[r];
  }
  __syncthreads();

  // ---- softmax: 2 rows per wave, float4 over 2048 cols ----
  const size_t obase = ((size_t)bh * SEQ + qb * 16) * SEQ;
  for (int rr = wave * 2; rr < wave * 2 + 2; ++rr) {
    float4* rowv = (float4*)(sP + rr * SEQ);
    float m = -3.4e38f;
    for (int c = lane; c < SEQ / 4; c += 32) {
      float4 v = rowv[c];
      m = fmaxf(m, fmaxf(fmaxf(v.x, v.y), fmaxf(v.z, v.w)));
    }
#pragma unroll
    for (int o = 16; o > 0; o >>= 1) m = fmaxf(m, __shfl_xor(m, o, 32));
    float s = 0.0f;
    for (int c = lane; c < SEQ / 4; c += 32) {
      float4 v = rowv[c];
      v.x = __expf(v.x - m); v.y = __expf(v.y - m);
      v.z = __expf(v.z - m); v.w = __expf(v.w - m);
      rowv[c] = v;
      s += (v.x + v.y) + (v.z + v.w);
    }
#pragma unroll
    for (int o = 16; o > 0; o >>= 1) s += __shfl_xor(s, o, 32);
    const float inv = 1.0f / s;
    float4* gout = (float4*)(attn_out + obase + (size_t)rr * SEQ);
    for (int c = lane; c < SEQ / 4; c += 32) {
      float4 v = rowv[c];
      v.x *= inv; v.y *= inv; v.z *= inv; v.w *= inv;
      rowv[c] = v;
      gout[c] = v;        // single coalesced b128 HBM write of attn
    }
  }
  __syncthreads();

  // ---- ctx = P(16x2048) @ V(2048x64): all 8 waves; 2 waves per 16-wide
  //      d tile, each covering half of K; combine partials through LDS ----
  const int nt  = wave & 3;           // d tile
  const int kh2 = wave >> 2;          // K half (0 or 1)
  const bf16* vbase = vt + ((size_t)bh * DK + nt * 16 + frow) * SEQ;
  v8f acc = {};
  const int kbeg = kh2 * (SEQ / 2);
  for (int k0 = kbeg; k0 < kbeg + SEQ / 2; k0 += 32) {
    const float* pp = sP + frow * SEQ + k0 + fk;
    float4 f0 = *(const float4*)(pp + 0);
    float4 f1 = *(const float4*)(pp + 4);
    float4 f2 = *(const float4*)(pp + 8);
    float4 f3 = *(const float4*)(pp + 12);
    v16bf a;
    a[0]=(bf16)f0.x;  a[1]=(bf16)f0.y;  a[2]=(bf16)f0.z;  a[3]=(bf16)f0.w;
    a[4]=(bf16)f1.x;  a[5]=(bf16)f1.y;  a[6]=(bf16)f1.z;  a[7]=(bf16)f1.w;
    a[8]=(bf16)f2.x;  a[9]=(bf16)f2.y;  a[10]=(bf16)f2.z; a[11]=(bf16)f2.w;
    a[12]=(bf16)f3.x; a[13]=(bf16)f3.y; a[14]=(bf16)f3.z; a[15]=(bf16)f3.w;
    v16bf bfr = *(const v16bf*)(vbase + k0 + fk);
    acc = wmma_bf16(a, bfr, acc);
  }
  float* sRed = sBias;                // dead after scores phase; 1024 floats
  if (kh2 == 1) {
#pragma unroll
    for (int r = 0; r < 8; ++r) sRed[nt * 256 + r * 32 + lane] = acc[r];
  }
  __syncthreads();
  if (kh2 == 0) {
#pragma unroll
    for (int r = 0; r < 8; ++r) {
      const float v = acc[r] + sRed[nt * 256 + r * 32 + lane];
      const int srow = qb * 16 + rbase + r;
      const int d    = h * DK + nt * 16 + frow;
      ctx[((size_t)b * SEQ + srow) * D_MODEL + d] = (bf16)v;
    }
  }
}

// ---------------------------------------------------------------------------
extern "C" void kernel_launch(void* const* d_in, const int* in_sizes, int n_in,
                              void* d_out, int out_size, void* d_ws, size_t ws_size,
                              hipStream_t stream) {
  (void)in_sizes; (void)n_in; (void)out_size; (void)ws_size;
  const float* query = (const float*)d_in[0];
  const float* key_  = (const float*)d_in[1];
  const float* value = (const float*)d_in[2];
  const float* Wq    = (const float*)d_in[3];
  const float* Wk    = (const float*)d_in[4];
  const float* Wv    = (const float*)d_in[5];
  const float* Wo    = (const float*)d_in[6];
  const float* bo    = (const float*)d_in[7];
  const float* rel   = (const float*)d_in[8];

  float* out_f  = (float*)d_out;                                   // (B,S,D)
  float* attn_f = out_f + (size_t)BATCH * SEQ * D_MODEL;           // (B,H,S,S)

  const size_t XN = (size_t)BATCH * SEQ * D_MODEL;  // 4M elements
  const size_t WN = (size_t)D_MODEL * D_MODEL;      // 1M elements
  bf16* p  = (bf16*)d_ws;
  bf16* xq = p; p += XN;
  bf16* xk = p; p += XN;
  bf16* xv = p; p += XN;
  bf16* wq = p; p += WN;
  bf16* wk = p; p += WN;
  bf16* wv = p; p += WN;
  bf16* wo = p; p += WN;
  bf16* qp = p; p += XN;   // projection outputs (B,S,D)
  bf16* kp = p; p += XN;
  bf16* vp = p; p += XN;
  bf16* qhd = p; p += XN;  // (B,H,S,dk) post-RoPE (q pre-scaled by 1/8)
  bf16* khd = p; p += XN;  // (B,H,S,dk) post-RoPE
  bf16* vtd = p; p += XN;  // (B,H,dk,S)
  bf16* cx  = p; p += XN;  // ctx (B,S,D)

  auto cvt = [&](const float* s, bf16* d, size_t n) {
    cvt_f32_bf16<<<(unsigned)((n / 4 + 255) / 256), 256, 0, stream>>>(s, d, (int)n);
  };
  cvt(query, xq, XN); cvt(key_, xk, XN); cvt(value, xv, XN);
  cvt(Wq, wq, WN); cvt(Wk, wk, WN); cvt(Wv, wv, WN); cvt(Wo, wo, WN);

  const int M = BATCH * SEQ, N = D_MODEL, K = D_MODEL;
  dim3 gg(N / 128, M / 64);   // (8, 64)
  gemm_bf16_nt<false, false><<<gg, 256, 0, stream>>>(xq, wq, nullptr, nullptr, qp, M, N, K);
  gemm_bf16_nt<false, false><<<gg, 256, 0, stream>>>(xk, wk, nullptr, nullptr, kp, M, N, K);
  gemm_bf16_nt<false, false><<<gg, 256, 0, stream>>>(xv, wv, nullptr, nullptr, vp, M, N, K);

  const size_t rope_threads = (size_t)BATCH * SEQ * N_HEADS * 32;  // 2^21
  rope_reshape<<<(unsigned)(rope_threads / 256), 256, 0, stream>>>(qp, qhd, 0.125f);
  rope_reshape<<<(unsigned)(rope_threads / 256), 256, 0, stream>>>(kp, khd, 1.0f);
  v_transpose<<<(unsigned)(XN / 256), 256, 0, stream>>>(vp, vtd);

  const size_t shmem = (size_t)16 * SEQ * 4 + 2080 * 4 + 16 * DK * 2;  // ~138 KB
  attn_fused<<<BATCH * N_HEADS * (SEQ / 16), 256, shmem, stream>>>(qhd, khd, vtd, rel,
                                                                   attn_f, cx);

  gemm_bf16_nt<true, true><<<gg, 256, 0, stream>>>(cx, wo, bo, out_f, nullptr, M, N, K);
}
